// EdgeAwareGATFusion_55722905698621
// MI455X (gfx1250) — matrix operand order repaired and checked by reading, not compiled, not run on hardware
//
#include <hip/hip_runtime.h>
#include <hip/hip_bf16.h>

#define N_NODES 50000
#define E_EDGES 400000
#define DIM     128
#define DFF_DIM 512

typedef __attribute__((ext_vector_type(16))) _Float16 v16h;
typedef __attribute__((ext_vector_type(8)))  _Float16 v8h;
typedef __attribute__((ext_vector_type(4)))  _Float16 v4h;
typedef __attribute__((ext_vector_type(8)))  float    v8f;

#define DEVINL __device__ __forceinline__

// within-wave LDS producer->consumer ordering (lanes are lockstep in a wave32;
// we only need the DS counter drained before dependent cross-lane LDS reads).
DEVINL void wave_lds_sync() { asm volatile("s_wait_dscnt 0" ::: "memory"); }

DEVINL float red16(float v) {
  v += __shfl_xor(v, 1, 32);
  v += __shfl_xor(v, 2, 32);
  v += __shfl_xor(v, 4, 32);
  v += __shfl_xor(v, 8, 32);
  return v;
}

DEVINL v8f wmma_f16(v16h a, v16h b, v8f c) {
  return __builtin_amdgcn_wmma_f32_16x16x32_f16(false, a, false, b, (short)0, c, false, false);
}

// A fragment (16x32 f16) from an LDS tile with row stride `stride` halves.
// ISA layout: lane = (m) + 16*((k%16)/8), element e -> K = (e>>3)*16 + g*8 + (e&7)
DEVINL v16h load_a_lds(const _Float16* t, int stride, int kbase, int lane) {
  const int g = (lane >> 4) & 1, m = lane & 15;
  const _Float16* p = t + m * stride + kbase + g * 8;
  v8h lo = *(const v8h*)(p);        // K = kbase + g*8 + [0..7]
  v8h hi = *(const v8h*)(p + 16);   // K = kbase + 16 + g*8 + [0..7]
  return __builtin_shufflevector(lo, hi, 0,1,2,3,4,5,6,7,8,9,10,11,12,13,14,15);
}

// B fragment (32x16 f16): lane holds col N = nbase+(lane&15), element e -> K = (e>>3)*16 + g*8 + (e&7).
DEVINL v16h load_b_glb(const _Float16* w, int ncols, int kbase, int nbase, int lane) {
  const int g = (lane >> 4) & 1, n = lane & 15;
  const _Float16* p = w + (size_t)(kbase + g * 8) * ncols + nbase + n;
  v16h b;
#pragma unroll
  for (int s = 0; s < 8; ++s) {
    b[s] = p[(size_t)s * ncols];
    b[s + 8] = p[(size_t)(16 + s) * ncols];
  }
  return b;
}

// per-row mean / rstd over the 128-wide row in WMMA C layout:
// lane l, reg r  ->  row = r + 8*(l>>4), col = ntile*16 + (l&15)
DEVINL void row_stats(const v8f* acc, float* mean, float* rstd) {
#pragma unroll
  for (int r = 0; r < 8; ++r) {
    float s = 0.f, s2 = 0.f;
#pragma unroll
    for (int n = 0; n < 8; ++n) { float v = acc[n][r]; s += v; s2 += v * v; }
    s = red16(s); s2 = red16(s2);
    float m = s * (1.f / 128.f);
    float var = s2 * (1.f / 128.f) - m * m;
    mean[r] = m;
    rstd[r] = rsqrtf(var + 1e-5f);
  }
}

// monotonic order-preserving float<->uint encoding for atomicMax
DEVINL unsigned encf(float x) {
  unsigned u = __float_as_uint(x);
  return (u & 0x80000000u) ? ~u : (u | 0x80000000u);
}
DEVINL float decf(unsigned e) {
  return __uint_as_float((e & 0x80000000u) ? (e ^ 0x80000000u) : ~e);
}

// ---------------------------------------------------------------- weights f16
__global__ void cvt_f16_kernel(const float* __restrict__ s, _Float16* __restrict__ d, int n) {
  int i = blockIdx.x * blockDim.x + threadIdx.x;
  if (i < n) d[i] = (_Float16)s[i];
}

// ---------------------------------------------------------------- edge kernel
__global__ __launch_bounds__(64)
void edge_kernel(const float* __restrict__ x, const int* __restrict__ ei,
                 const float* __restrict__ eattr,
                 const _Float16* __restrict__ wmemh, const float* __restrict__ bmem,
                 const float* __restrict__ gmem, const float* __restrict__ bbmem,
                 const _Float16* __restrict__ wqh, const _Float16* __restrict__ wkh,
                 const _Float16* __restrict__ wvh, const _Float16* __restrict__ weuh,
                 const float* __restrict__ beu, const float* __restrict__ geu,
                 const float* __restrict__ bbeu, const float* __restrict__ gen,
                 const float* __restrict__ bben,
                 _Float16* __restrict__ vbuf, float* __restrict__ logits,
                 unsigned* __restrict__ nmax, float* __restrict__ edge_out)
{
  __shared__ __align__(16) _Float16 s_concat[2][16 * 384];   // [xi | xj | ea] f16
  __shared__ __align__(16) _Float16 s_mem[2][16 * 128];      // mem tile f16
  __shared__ int s_src[2][16], s_tgt[2][16];

  const int wave = threadIdx.x >> 5;
  const int lane = threadIdx.x & 31;
  const int tile = blockIdx.x * 2 + wave;
  const int base = tile * 16;
  _Float16* conc = s_concat[wave];
  _Float16* memT = s_mem[wave];
  int* stgt = s_tgt[wave];
  int* ssrc = s_src[wave];
  const int g = lane >> 4, lm = lane & 15;

  if (lane < 16) {
    ssrc[lane] = ei[base + lane];
    stgt[lane] = ei[E_EDGES + base + lane];
  }
  wave_lds_sync();

  // gather rows + convert to f16  (one float4 per lane covers 128 cols)
  for (int r = 0; r < 16; ++r) {
    int ti = stgt[r], si = ssrc[r];
    float4 a = *(const float4*)(x + (size_t)ti * DIM + lane * 4);
    float4 b = *(const float4*)(x + (size_t)si * DIM + lane * 4);
    float4 c = *(const float4*)(eattr + (size_t)(base + r) * DIM + lane * 4);
    v4h ha = {(_Float16)a.x, (_Float16)a.y, (_Float16)a.z, (_Float16)a.w};
    v4h hb = {(_Float16)b.x, (_Float16)b.y, (_Float16)b.z, (_Float16)b.w};
    v4h hc = {(_Float16)c.x, (_Float16)c.y, (_Float16)c.z, (_Float16)c.w};
    *(v4h*)(conc + r * 384 + lane * 4) = ha;
    *(v4h*)(conc + r * 384 + 128 + lane * 4) = hb;
    *(v4h*)(conc + r * 384 + 256 + lane * 4) = hc;
  }
  wave_lds_sync();

  float mean[8], rstd[8];

  // ---- GEMM1: mem_pre = concat @ W_mem  (K=384 -> 12 k-steps, 8 n-tiles)
  v8f acc[8];
#pragma unroll
  for (int n = 0; n < 8; ++n) acc[n] = v8f{};
  for (int k = 0; k < 12; ++k) {
    v16h a = load_a_lds(conc, 384, k * 32, lane);
#pragma unroll
    for (int n = 0; n < 8; ++n)
      acc[n] = wmma_f16(a, load_b_glb(wmemh, 128, k * 32, n * 16, lane), acc[n]);
  }
#pragma unroll
  for (int n = 0; n < 8; ++n) {
    float bv = bmem[n * 16 + lm];
#pragma unroll
    for (int r = 0; r < 8; ++r) acc[n][r] += bv;
  }
  row_stats(acc, mean, rstd);
#pragma unroll
  for (int n = 0; n < 8; ++n) {
    float gv = gmem[n * 16 + lm], bv = bbmem[n * 16 + lm];
#pragma unroll
    for (int r = 0; r < 8; ++r) {
      float v = (acc[n][r] - mean[r]) * rstd[r] * gv + bv;
      v = v > 0.f ? v : 0.f;                       // ReLU
      memT[(r + 8 * g) * 128 + n * 16 + lm] = (_Float16)v;
    }
  }
  wave_lds_sync();

  // ---- q = x_i @ Wq   (x_i = concat cols 0..127)
  v8f qa[8], ka[8];
#pragma unroll
  for (int n = 0; n < 8; ++n) { qa[n] = v8f{}; ka[n] = v8f{}; }
  for (int k = 0; k < 4; ++k) {
    v16h a = load_a_lds(conc, 384, k * 32, lane);
#pragma unroll
    for (int n = 0; n < 8; ++n)
      qa[n] = wmma_f16(a, load_b_glb(wqh, 128, k * 32, n * 16, lane), qa[n]);
  }
  // ---- k = mem @ Wk
  for (int k = 0; k < 4; ++k) {
    v16h a = load_a_lds(memT, 128, k * 32, lane);
#pragma unroll
    for (int n = 0; n < 8; ++n)
      ka[n] = wmma_f16(a, load_b_glb(wkh, 128, k * 32, n * 16, lane), ka[n]);
  }

  // ---- logits[row][h] = (q . k)/sqrt(16); n-tile == head
  float lg[8];
#pragma unroll
  for (int r = 0; r < 8; ++r) lg[r] = 0.f;
#pragma unroll
  for (int r = 0; r < 8; ++r) {
#pragma unroll
    for (int n = 0; n < 8; ++n) {
      float red = red16(qa[n][r] * ka[n][r]) * 0.25f;
      if (lm == n) lg[r] = red;
    }
  }
  if (lm < 8) {
#pragma unroll
    for (int r = 0; r < 8; ++r) {
      int row = r + 8 * g;
      int e = base + row;
      logits[(size_t)e * 8 + lm] = lg[r];
      atomicMax(nmax + (size_t)stgt[row] * 8 + lm, encf(lg[r]));
    }
  }

  // ---- v = mem @ Wv  -> store f16 to vbuf (stage in conc cols 0..127)
  v8f va[8];
#pragma unroll
  for (int n = 0; n < 8; ++n) va[n] = v8f{};
  for (int k = 0; k < 4; ++k) {
    v16h a = load_a_lds(memT, 128, k * 32, lane);
#pragma unroll
    for (int n = 0; n < 8; ++n)
      va[n] = wmma_f16(a, load_b_glb(wvh, 128, k * 32, n * 16, lane), va[n]);
  }
#pragma unroll
  for (int n = 0; n < 8; ++n)
#pragma unroll
    for (int r = 0; r < 8; ++r)
      conc[(r + 8 * g) * 384 + n * 16 + lm] = (_Float16)va[n][r];
  wave_lds_sync();
#pragma unroll
  for (int i = 0; i < 8; ++i) {
    int c = i * 32 + lane;              // 256 chunks of 8 halves
    int row = c >> 4, off = (c & 15) * 8;
    uint4 d = *(const uint4*)(conc + row * 384 + off);
    *(uint4*)(vbuf + (size_t)(base + row) * 128 + off) = d;
  }
  wave_lds_sync();

  // ---- edge update: delta = ReLU(LN(mem @ W_eu + b_eu)); edge_out = LN(ea + delta)
  v8f ua[8];
#pragma unroll
  for (int n = 0; n < 8; ++n) ua[n] = v8f{};
  for (int k = 0; k < 4; ++k) {
    v16h a = load_a_lds(memT, 128, k * 32, lane);
#pragma unroll
    for (int n = 0; n < 8; ++n)
      ua[n] = wmma_f16(a, load_b_glb(weuh, 128, k * 32, n * 16, lane), ua[n]);
  }
#pragma unroll
  for (int n = 0; n < 8; ++n) {
    float bv = beu[n * 16 + lm];
#pragma unroll
    for (int r = 0; r < 8; ++r) ua[n][r] += bv;
  }
  row_stats(ua, mean, rstd);
  v8f eo[8];
#pragma unroll
  for (int n = 0; n < 8; ++n) {
    float gv = geu[n * 16 + lm], bv = bbeu[n * 16 + lm];
#pragma unroll
    for (int r = 0; r < 8; ++r) {
      float d = (ua[n][r] - mean[r]) * rstd[r] * gv + bv;
      d = d > 0.f ? d : 0.f;
      float eav = (float)conc[(r + 8 * g) * 384 + 256 + n * 16 + lm];
      eo[n][r] = eav + d;
    }
  }
  row_stats(eo, mean, rstd);
#pragma unroll
  for (int n = 0; n < 8; ++n) {
    float gv = gen[n * 16 + lm], bv = bben[n * 16 + lm];
#pragma unroll
    for (int r = 0; r < 8; ++r) {
      float o = (eo[n][r] - mean[r]) * rstd[r] * gv + bv;
      ((float*)(conc + (r + 8 * g) * 384))[n * 16 + lm] = o;  // f32 stage (cols 0..255)
    }
  }
  wave_lds_sync();
#pragma unroll
  for (int i = 0; i < 16; ++i) {
    int c = i * 32 + lane;              // 512 float4 chunks
    int row = c >> 5, off = (c & 31) * 4;
    float4 d = *(const float4*)((const float*)(conc + row * 384) + off);
    *(float4*)(edge_out + (size_t)(base + row) * 128 + off) = d;
  }
}

// ------------------------------------------------------------- softmax passes
__global__ void softmax_pass(const int* __restrict__ ei, const float* __restrict__ logits,
                             const unsigned* __restrict__ nmax,
                             float* __restrict__ eexp, float* __restrict__ nsum)
{
  int t = blockIdx.x * blockDim.x + threadIdx.x;
  if (t >= E_EDGES * 8) return;
  int e = t >> 3, h = t & 7;
  int tg = ei[E_EDGES + e];
  float m = decf(nmax[(size_t)tg * 8 + h]);
  float ex = __expf(logits[t] - m);
  eexp[t] = ex;
  atomicAdd(nsum + (size_t)tg * 8 + h, ex);
}

__global__ void aggregate_pass(const int* __restrict__ ei, const float* __restrict__ eexp,
                               const float* __restrict__ nsum,
                               const _Float16* __restrict__ vbuf, float* __restrict__ aggr)
{
  int t = blockIdx.x * blockDim.x + threadIdx.x;   // E*128 = 51.2M < 2^31
  int e = t >> 7, d = t & 127, h = d >> 4;
  int tg = ei[E_EDGES + e];
  float a = eexp[(size_t)e * 8 + h] / (nsum[(size_t)tg * 8 + h] + 1e-16f);
  atomicAdd(aggr + (size_t)tg * 128 + d, a * (float)vbuf[(size_t)e * 128 + d]);
}

// ---------------------------------------------------------------- node kernel
__global__ __launch_bounds__(64)
void node_kernel(const float* __restrict__ x, const float* __restrict__ aggr,
                 const _Float16* __restrict__ woh,
                 const float* __restrict__ g1, const float* __restrict__ bb1,
                 const _Float16* __restrict__ w1h, const float* __restrict__ b1,
                 const _Float16* __restrict__ w2h, const float* __restrict__ b2,
                 const float* __restrict__ g2, const float* __restrict__ bb2,
                 float* __restrict__ out)
{
  __shared__ __align__(16) float    s_x[2][16 * 128];
  __shared__ __align__(16) _Float16 s_a[2][16 * 128];   // aggr f16, then h f16
  __shared__ __align__(16) _Float16 s_f[2][16 * 512];   // ffn1 activations

  const int wave = threadIdx.x >> 5;
  const int lane = threadIdx.x & 31;
  const int tile = blockIdx.x * 2 + wave;
  if (tile >= N_NODES / 16) return;
  const int base = tile * 16;
  float* xs = s_x[wave];
  _Float16* as = s_a[wave];
  _Float16* fs = s_f[wave];
  const int g = lane >> 4, lm = lane & 15;

  // stage x (f32) and aggr (f16)
#pragma unroll
  for (int i = 0; i < 16; ++i) {
    float4 xv = *(const float4*)(x + (size_t)base * 128 + (i * 32 + lane) * 4);
    *(float4*)(xs + (i * 32 + lane) * 4) = xv;
    float4 av = *(const float4*)(aggr + (size_t)base * 128 + (i * 32 + lane) * 4);
    v4h h = {(_Float16)av.x, (_Float16)av.y, (_Float16)av.z, (_Float16)av.w};
    *(v4h*)(as + (i * 32 + lane) * 4) = h;
  }
  wave_lds_sync();

  float mean[8], rstd[8];

  // ---- aggr @ Wo
  v8f acc[8];
#pragma unroll
  for (int n = 0; n < 8; ++n) acc[n] = v8f{};
  for (int k = 0; k < 4; ++k) {
    v16h a = load_a_lds(as, 128, k * 32, lane);
#pragma unroll
    for (int n = 0; n < 8; ++n)
      acc[n] = wmma_f16(a, load_b_glb(woh, 128, k * 32, n * 16, lane), acc[n]);
  }
  // ---- h = LN(x + aggrO)
#pragma unroll
  for (int n = 0; n < 8; ++n)
#pragma unroll
    for (int r = 0; r < 8; ++r)
      acc[n][r] += xs[(r + 8 * g) * 128 + n * 16 + lm];
  row_stats(acc, mean, rstd);
  v8f h8[8];
#pragma unroll
  for (int n = 0; n < 8; ++n) {
    float gv = g1[n * 16 + lm], bv = bb1[n * 16 + lm];
#pragma unroll
    for (int r = 0; r < 8; ++r) {
      float hv = (acc[n][r] - mean[r]) * rstd[r] * gv + bv;
      h8[n][r] = hv;
      as[(r + 8 * g) * 128 + n * 16 + lm] = (_Float16)hv;
    }
  }
  wave_lds_sync();

  // ---- FFN1: relu(h @ W1 + b1) -> s_f  (A fragments hoisted over 32 n-tiles)
  v16h afr[4];
#pragma unroll
  for (int k = 0; k < 4; ++k) afr[k] = load_a_lds(as, 128, k * 32, lane);
  for (int n2 = 0; n2 < 32; ++n2) {
    v8f a1 = v8f{};
#pragma unroll
    for (int k = 0; k < 4; ++k)
      a1 = wmma_f16(afr[k], load_b_glb(w1h, 512, k * 32, n2 * 16, lane), a1);
    float bv = b1[n2 * 16 + lm];
#pragma unroll
    for (int r = 0; r < 8; ++r) {
      float vv = a1[r] + bv;
      vv = vv > 0.f ? vv : 0.f;
      fs[(r + 8 * g) * 512 + n2 * 16 + lm] = (_Float16)vv;
    }
  }
  wave_lds_sync();

  // ---- FFN2: ffn1 @ W2
  v8f f2[8];
#pragma unroll
  for (int n = 0; n < 8; ++n) f2[n] = v8f{};
  for (int k = 0; k < 16; ++k) {
    v16h a = load_a_lds(fs, 512, k * 32, lane);
#pragma unroll
    for (int n = 0; n < 8; ++n)
      f2[n] = wmma_f16(a, load_b_glb(w2h, 128, k * 32, n * 16, lane), f2[n]);
  }
  // ---- out = LN(h + ffn2 + b2)
#pragma unroll
  for (int n = 0; n < 8; ++n) {
    float bv = b2[n * 16 + lm];
#pragma unroll
    for (int r = 0; r < 8; ++r) f2[n][r] = h8[n][r] + f2[n][r] + bv;
  }
  row_stats(f2, mean, rstd);
#pragma unroll
  for (int n = 0; n < 8; ++n) {
    float gv = g2[n * 16 + lm], bv = bb2[n * 16 + lm];
#pragma unroll
    for (int r = 0; r < 8; ++r)
      xs[(r + 8 * g) * 128 + n * 16 + lm] = (f2[n][r] - mean[r]) * rstd[r] * gv + bv;
  }
  wave_lds_sync();
#pragma unroll
  for (int i = 0; i < 16; ++i)
    *(float4*)(out + (size_t)base * 128 + (i * 32 + lane) * 4) =
        *(const float4*)(xs + (i * 32 + lane) * 4);
}

// ------------------------------------------------------------------ launcher
extern "C" void kernel_launch(void* const* d_in, const int* in_sizes, int n_in,
                              void* d_out, int out_size, void* d_ws, size_t ws_size,
                              hipStream_t stream) {
  const float* x      = (const float*)d_in[0];
  const int*   ei     = (const int*)d_in[1];
  const float* eattr  = (const float*)d_in[2];
  const float* W_mem  = (const float*)d_in[3];
  const float* b_mem  = (const float*)d_in[4];
  const float* g_mem  = (const float*)d_in[5];
  const float* bb_mem = (const float*)d_in[6];
  const float* Wq     = (const float*)d_in[7];
  const float* Wk     = (const float*)d_in[8];
  const float* Wv     = (const float*)d_in[9];
  const float* Wo     = (const float*)d_in[10];
  const float* W_eu   = (const float*)d_in[11];
  const float* b_eu   = (const float*)d_in[12];
  const float* g_eu   = (const float*)d_in[13];
  const float* bb_eu  = (const float*)d_in[14];
  const float* g_en   = (const float*)d_in[15];
  const float* bb_en  = (const float*)d_in[16];
  const float* W1     = (const float*)d_in[17];
  const float* b1     = (const float*)d_in[18];
  const float* W2     = (const float*)d_in[19];
  const float* b2     = (const float*)d_in[20];
  const float* g1     = (const float*)d_in[21];
  const float* bb1    = (const float*)d_in[22];
  const float* g2     = (const float*)d_in[23];
  const float* bb2    = (const float*)d_in[24];

  // workspace layout
  constexpr int WMEM_H = 0, WQ_H = 49152, WK_H = 65536, WV_H = 81920,
                WO_H = 98304, WEU_H = 114688, W1_H = 131072, W2_H = 196608;
  constexpr size_t WH_BYTES = 524288;                                  // 262144 halves
  constexpr size_t VBUF_OFF = WH_BYTES;                                // E*128 f16
  constexpr size_t LOG_OFF  = VBUF_OFF + (size_t)E_EDGES * 128 * 2;    // E*8 f32
  constexpr size_t EXP_OFF  = LOG_OFF + (size_t)E_EDGES * 8 * 4;       // E*8 f32
  constexpr size_t NMAX_OFF = EXP_OFF + (size_t)E_EDGES * 8 * 4;       // N*8 u32
  constexpr size_t NSUM_OFF = NMAX_OFF + (size_t)N_NODES * 8 * 4;      // N*8 f32
  constexpr size_t AGGR_OFF = NSUM_OFF + (size_t)N_NODES * 8 * 4;      // N*128 f32

  char* ws = (char*)d_ws;
  _Float16* wh    = (_Float16*)(ws);
  _Float16* vbuf  = (_Float16*)(ws + VBUF_OFF);
  float*    logb  = (float*)(ws + LOG_OFF);
  float*    eexp  = (float*)(ws + EXP_OFF);
  unsigned* nmax  = (unsigned*)(ws + NMAX_OFF);
  float*    nsum  = (float*)(ws + NSUM_OFF);
  float*    aggr  = (float*)(ws + AGGR_OFF);

  float* out_node = (float*)d_out;
  float* out_edge = (float*)d_out + (size_t)N_NODES * 128;

  // 0) convert weights to f16 (every call; deterministic)
  auto cvt = [&](const float* s, _Float16* d, int n) {
    cvt_f16_kernel<<<(n + 255) / 256, 256, 0, stream>>>(s, d, n);
  };
  cvt(W_mem, wh + WMEM_H, 49152);
  cvt(Wq,    wh + WQ_H,   16384);
  cvt(Wk,    wh + WK_H,   16384);
  cvt(Wv,    wh + WV_H,   16384);
  cvt(Wo,    wh + WO_H,   16384);
  cvt(W_eu,  wh + WEU_H,  16384);
  cvt(W1,    wh + W1_H,   65536);
  cvt(W2,    wh + W2_H,   65536);

  // zero nmax (encoded -inf == 0), nsum, aggr in one contiguous memset
  hipMemsetAsync(ws + NMAX_OFF, 0,
                 (AGGR_OFF - NMAX_OFF) + (size_t)N_NODES * 128 * 4, stream);

  // 1) edge pipeline: 25000 tiles, 2 waves/block
  edge_kernel<<<E_EDGES / 16 / 2, 64, 0, stream>>>(
      x, ei, eattr, wh + WMEM_H, b_mem, g_mem, bb_mem,
      wh + WQ_H, wh + WK_H, wh + WV_H, wh + WEU_H,
      b_eu, g_eu, bb_eu, g_en, bb_en,
      vbuf, logb, nmax, out_edge);

  // 2) segment softmax numerator + denominator
  softmax_pass<<<(E_EDGES * 8 + 255) / 256, 256, 0, stream>>>(ei, logb, nmax, eexp, nsum);

  // 3) weighted scatter-add of v
  aggregate_pass<<<(E_EDGES * 128) / 256, 256, 0, stream>>>(ei, eexp, nsum, vbuf, aggr);

  // 4) node pipeline: 3125 tiles, 2 waves/block
  node_kernel<<<(N_NODES / 16 + 1) / 2, 64, 0, stream>>>(
      x, aggr, wh + WO_H, g1, bb1, wh + W1_H, b1, wh + W2_H, b2, g2, bb2, out_node);
}
